// Decoder_17506286698900
// MI455X (gfx1250) — matrix operand order repaired and checked by reading, not compiled, not run on hardware
//
#include <hip/hip_runtime.h>
#include <math.h>

// ---------- types ----------
typedef __attribute__((ext_vector_type(16))) __bf16 v16bf;
typedef __attribute__((ext_vector_type(8)))  __bf16 v8bf;
typedef __attribute__((ext_vector_type(4)))  __bf16 v4bf;
typedef __attribute__((ext_vector_type(8)))  float  v8f;
typedef __attribute__((ext_vector_type(4)))  float  v4f;

// ---------- problem dims ----------
#define B_N    64
#define E_DIM  256
#define H_DIM  512
#define H2     1024      // 2H
#define L_DIM  1024
#define V_DIM  50000
#define EXT_V  (V_DIM + L_DIM)   // 51024
#define K_LSTM 1792              // E + 2H (w_ih) + H (w_hh)
#define K_VOC  1536              // 3H

// ---------- d_out layout (floats, reference tuple order) ----------
#define O_FINAL 0
#define O_H     (B_N * EXT_V)
#define O_C     (O_H + B_N * H_DIM)
#define O_CTX   (O_C + B_N * H_DIM)
#define O_ATTN  (O_CTX + B_N * H2)
#define O_PGEN  (O_ATTN + B_N * L_DIM)
#define O_COV   (O_PGEN + B_N)

// ---------- workspace layout (floats) ----------
#define W_GATES 0
#define W_DECF  (W_GATES + B_N * 4 * H_DIM)
#define W_E     (W_DECF + B_N * H2)
#define W_PG    (W_E + B_N * L_DIM)
#define W_MAX   (W_PG + B_N)
#define W_SUM   (W_MAX + B_N)

// ---------- WMMA helper: D = A(16x32 bf16) x B(32x16 bf16) + C(f32) ----------
__device__ __forceinline__ v8f wmma_bf16(v16bf a, v16bf b, v8f c) {
  return __builtin_amdgcn_wmma_f32_16x16x32_bf16(
      /*neg_a=*/false, a, /*neg_b=*/false, b,
      /*c_mod=*/(short)0, c, /*reuse_a=*/false, /*reuse_b=*/false);
}

// 8 consecutive f32 -> bf16 into frag elems [o..o+7]
__device__ __forceinline__ void cvt8(v16bf& d, int o, const float* __restrict__ p) {
  #pragma unroll
  for (int i = 0; i < 8; ++i) d[o + i] = (__bf16)p[i];
}
// Non-temporal flavor (streaming weights/activations)
__device__ __forceinline__ void cvt8_nt(v16bf& d, int o, const float* __restrict__ p) {
  v4f x = __builtin_nontemporal_load((const v4f*)p);
  v4f y = __builtin_nontemporal_load((const v4f*)(p + 4));
  d[o + 0] = (__bf16)x[0]; d[o + 1] = (__bf16)x[1];
  d[o + 2] = (__bf16)x[2]; d[o + 3] = (__bf16)x[3];
  d[o + 4] = (__bf16)y[0]; d[o + 5] = (__bf16)y[1];
  d[o + 6] = (__bf16)y[2]; d[o + 7] = (__bf16)y[3];
}

// =====================================================================
// Kernel 1: LSTM gates GEMM: gates(64,2048) = [emb|ctx|h0] @ [w_ih|w_hh]^T + b
// Branch-free (K split at 256/1280); dual accumulators break the
// WMMA->WMMA RAW hazard chain (no v_nop slots).
// =====================================================================
__global__ __launch_bounds__(64)
void k_gates(const int* __restrict__ tok, const float* __restrict__ emb,
             const float* __restrict__ ctx, const float* __restrict__ h0,
             const float* __restrict__ w_ih, const float* __restrict__ w_hh,
             const float* __restrict__ b_ih, const float* __restrict__ b_hh,
             float* __restrict__ gates) {
  const int lane  = threadIdx.x & 31;
  const int wave  = threadIdx.x >> 5;
  const int wid   = blockIdx.x * 2 + wave;   // 0..511
  const int ntile = wid >> 2;                // 0..127
  const int mtile = wid & 3;                 // 0..3
  const int n     = ntile * 16 + (lane & 15);
  const int mrow  = mtile * 16 + (lane & 15);
  const int hi8   = (lane >> 4) * 8;
  const int koff  = hi8;                     // A frag layout: 0 or 8
  const int kb16  = (lane >> 4) * 16;        // unused; B frag also uses koff (0 or 8)

  // hoisted row base pointers (biased so that ptr + k0 is correct)
  const float* __restrict__ a_emb = emb + (size_t)tok[mrow] * E_DIM;
  const float* __restrict__ a_ctx = ctx + (size_t)mrow * H2 - E_DIM;
  const float* __restrict__ a_h0  = h0  + (size_t)mrow * H_DIM - (E_DIM + H2);
  const float* __restrict__ b_ih_row = w_ih + (size_t)n * (E_DIM + H2);
  const float* __restrict__ b_hh_row = w_hh + (size_t)n * H_DIM - (E_DIM + H2);

  v8f acc0 = {}, acc1 = {};
  auto step = [&](const float* __restrict__ Ab, const float* __restrict__ Bb,
                  int k0, int par) {
    v16bf a, b;
    cvt8(a, 0, Ab + k0 + koff);
    cvt8(a, 8, Ab + k0 + 16 + koff);
    cvt8(b, 0, Bb + k0 + koff);
    cvt8(b, 8, Bb + k0 + 16 + koff);
    if (par) acc1 = wmma_bf16(a, b, acc1);
    else     acc0 = wmma_bf16(a, b, acc0);
  };

  #pragma unroll 4
  for (int k0 = 0; k0 < E_DIM; k0 += 32)            step(a_emb, b_ih_row, k0, (k0 >> 5) & 1);
  #pragma unroll 4
  for (int k0 = E_DIM; k0 < E_DIM + H2; k0 += 32)   step(a_ctx, b_ih_row, k0, (k0 >> 5) & 1);
  #pragma unroll 4
  for (int k0 = E_DIM + H2; k0 < K_LSTM; k0 += 32)  step(a_h0, b_hh_row, k0, (k0 >> 5) & 1);

  const float bias = b_ih[n] + b_hh[n];
  #pragma unroll
  for (int r = 0; r < 8; ++r) {
    const int m = mtile * 16 + r + hi8;
    gates[(size_t)m * (4 * H_DIM) + n] = acc0[r] + acc1[r] + bias;
  }
}

// =====================================================================
// Kernel 2: LSTM pointwise -> h_new, c_new (into d_out)
// =====================================================================
__global__ __launch_bounds__(256)
void k_lstm_point(const float* __restrict__ gates, const float* __restrict__ c0,
                  float* __restrict__ out) {
  const int idx = blockIdx.x * 256 + threadIdx.x;   // 0..32767
  const int b = idx >> 9, h = idx & 511;
  const float* g = gates + (size_t)b * 2048;
  const float gi = g[h], gf = g[512 + h], gg = g[1024 + h], go = g[1536 + h];
  const float si = 1.f / (1.f + expf(-gi));
  const float sf = 1.f / (1.f + expf(-gf));
  const float so = 1.f / (1.f + expf(-go));
  const float c  = sf * c0[idx] + si * tanhf(gg);
  out[O_C + idx] = c;
  out[O_H + idx] = so * tanhf(c);
}

// =====================================================================
// Kernel 3: dec_feat(64,1024) = h_new @ Ws^T   (tiny)
// =====================================================================
__global__ __launch_bounds__(256)
void k_decfeat(const float* __restrict__ hnew, const float* __restrict__ Ws,
               float* __restrict__ decf) {
  const int idx = blockIdx.x * 256 + threadIdx.x;   // 0..65535
  const int b = idx >> 10, d = idx & 1023;
  const float* hp = hnew + (size_t)b * H_DIM;
  const float* wp = Ws + (size_t)d * H_DIM;
  float s = 0.f;
  for (int k = 0; k < H_DIM; ++k) s += hp[k] * wp[k];
  decf[idx] = s;
}

// =====================================================================
// Kernel 4: fused attention scores.
// WG = 256 thr (8 waves), 16 rows of enc staged to LDS as bf16 (32 KB, NT loads).
// Wave w sweeps d-tiles {w, w+8, ...} (Wh stays L2-resident, RT hint),
// dual-accumulator WMMA, tanh/v epilogue, shuffle + ds_add_f32 reduction.
// =====================================================================
__global__ __launch_bounds__(256)
void k_attn_score(const float* __restrict__ enc, const float* __restrict__ Wh,
                  const float* __restrict__ decf, const float* __restrict__ cov,
                  const float* __restrict__ wc, const float* __restrict__ battn,
                  const float* __restrict__ vvec, const float* __restrict__ mask,
                  float* __restrict__ e_out) {
  __shared__ __attribute__((aligned(16))) __bf16 ldsA[16][H2];  // 32 KB
  __shared__ float ered[16];
  const int tid  = threadIdx.x;
  const int lane = tid & 31;
  const int wave = tid >> 5;
  const int strip = blockIdx.x * 16;     // base row in (B*L, 1024)
  const int b     = strip >> 10;         // strip never crosses a batch row
  const int lbase = strip & 1023;
  const int hi8   = (lane >> 4) * 8;

  // stage A strip (16 x 1024 f32 -> bf16), streaming (non-temporal) loads
  for (int t = tid; t < 16 * H2 / 4; t += 256) {
    const int r  = t >> 8;
    const int kc = (t & 255) * 4;
    v4f f = __builtin_nontemporal_load(
        (const v4f*)(enc + (size_t)(strip + r) * H2 + kc));
    v4bf h4;
    h4[0] = (__bf16)f[0]; h4[1] = (__bf16)f[1];
    h4[2] = (__bf16)f[2]; h4[3] = (__bf16)f[3];
    *(v4bf*)&ldsA[r][kc] = h4;
  }
  if (tid < 16) ered[tid] = 0.f;
  __syncthreads();

  const int arow = lane & 15;
  const int koff = hi8;                  // A frag layout
  const int nb   = lane & 15;            // B frag column
  const int kb16 = (lane >> 4) * 16;     // unused; B frag K base is hi8 (0 or 8)

  float covr[8];
  #pragma unroll
  for (int r = 0; r < 8; ++r) covr[r] = cov[b * L_DIM + lbase + r + hi8];

  float ep[8];
  #pragma unroll
  for (int r = 0; r < 8; ++r) ep[r] = 0.f;

  for (int dt = 0; dt < 8; ++dt) {
    const int d0 = (dt * 8 + wave) * 16;
    v8f acc0 = {}, acc1 = {};
    const float* __restrict__ Brow = Wh + (size_t)(d0 + nb) * H2 + hi8;
    #pragma unroll 8
    for (int kk = 0; kk < 32; ++kk) {
      v16bf a, bf;
      {
        v8bf lo = *(const v8bf*)(&ldsA[arow][kk * 32 + koff]);
        v8bf hi = *(const v8bf*)(&ldsA[arow][kk * 32 + 16 + koff]);
        #pragma unroll
        for (int i = 0; i < 8; ++i) { a[i] = lo[i]; a[8 + i] = hi[i]; }
      }
      cvt8(bf, 0, Brow + kk * 32);
      cvt8(bf, 8, Brow + kk * 32 + 16);
      if (kk & 1) acc1 = wmma_bf16(a, bf, acc1);
      else        acc0 = wmma_bf16(a, bf, acc0);
    }
    // epilogue: v[d] * tanh(acc + dec_feat + cov*wc + b_attn)
    const int d = d0 + nb;
    const float df  = decf[b * H2 + d] + battn[d];
    const float wcd = wc[d];
    const float vd  = vvec[d];
    #pragma unroll
    for (int r = 0; r < 8; ++r)
      ep[r] += vd * tanhf(acc0[r] + acc1[r] + df + covr[r] * wcd);
  }

  // reduce across 16 columns held in a lane-group, then across waves
  #pragma unroll
  for (int r = 0; r < 8; ++r) {
    float s = ep[r];
    s += __shfl_xor(s, 1);
    s += __shfl_xor(s, 2);
    s += __shfl_xor(s, 4);
    s += __shfl_xor(s, 8);
    if ((lane & 15) == 0) atomicAdd(&ered[r + hi8], s);   // ds_add_f32
  }
  __syncthreads();
  if (tid < 16) {
    const int lpos = lbase + tid;
    const float mv = mask[b * L_DIM + lpos];
    e_out[b * L_DIM + lpos] = (mv > 0.5f) ? ered[tid] : -1e9f;
  }
}

// =====================================================================
// Kernel 5: softmax over L; attn + cov_new -> d_out
// =====================================================================
__global__ __launch_bounds__(256)
void k_softmax_attn(const float* __restrict__ e, const float* __restrict__ cov,
                    float* __restrict__ out) {
  __shared__ float red[256];
  const int b = blockIdx.x, tid = threadIdx.x;
  const float* er = e + b * L_DIM;
  float m = -INFINITY;
  for (int j = tid; j < L_DIM; j += 256) m = fmaxf(m, er[j]);
  red[tid] = m; __syncthreads();
  for (int s = 128; s > 0; s >>= 1) {
    if (tid < s) red[tid] = fmaxf(red[tid], red[tid + s]);
    __syncthreads();
  }
  m = red[0]; __syncthreads();
  float sum = 0.f;
  for (int j = tid; j < L_DIM; j += 256) sum += expf(er[j] - m);
  red[tid] = sum; __syncthreads();
  for (int s = 128; s > 0; s >>= 1) {
    if (tid < s) red[tid] += red[tid + s];
    __syncthreads();
  }
  const float inv = 1.f / red[0];
  for (int j = tid; j < L_DIM; j += 256) {
    const float a = expf(er[j] - m) * inv;
    out[O_ATTN + b * L_DIM + j] = a;
    out[O_COV  + b * L_DIM + j] = cov[b * L_DIM + j] + a;
  }
}

// =====================================================================
// Kernel 6: ctx(64,1024) = attn @ enc  (streaming, NT loads)
// =====================================================================
__global__ __launch_bounds__(256)
void k_context(const float* __restrict__ attn, const float* __restrict__ enc,
               float* __restrict__ out) {
  const int b = blockIdx.x >> 2;
  const int k = ((blockIdx.x & 3) << 8) + threadIdx.x;  // 0..1023
  const float* ar = attn + b * L_DIM;
  const float* er = enc + (size_t)b * L_DIM * H2 + k;
  float s = 0.f;
  for (int l = 0; l < L_DIM; ++l)
    s += ar[l] * __builtin_nontemporal_load(er + (size_t)l * H2);
  out[O_CTX + b * H2 + k] = s;
}

// =====================================================================
// Kernel 7: vocab logits GEMM: (64,50000) = [h|ctx](64,1536) @ Wv^T + bv
// One wave per 16-column tile, all 4 M-tiles (4 independent acc chains);
// W_vocab streamed with NT loads; K split at 512 (branch-free body).
// =====================================================================
__global__ __launch_bounds__(64)
void k_vocab(const float* __restrict__ hnew, const float* __restrict__ ctx,
             const float* __restrict__ Wv, const float* __restrict__ bv,
             float* __restrict__ out) {
  const int lane  = threadIdx.x & 31;
  const int wave  = threadIdx.x >> 5;
  const int ntile = blockIdx.x * 2 + wave;
  if (ntile >= V_DIM / 16) return;
  const int n    = ntile * 16 + (lane & 15);
  const int hi8  = (lane >> 4) * 8;
  const int koff = hi8;
  const int kb16 = (lane >> 4) * 16;

  // hoisted per-mtile A row pointers (ctx biased by -512)
  const float* __restrict__ ah[4];
  const float* __restrict__ ac[4];
  #pragma unroll
  for (int mt = 0; mt < 4; ++mt) {
    const int m = mt * 16 + (lane & 15);
    ah[mt] = hnew + (size_t)m * H_DIM;
    ac[mt] = ctx + (size_t)m * H2 - H_DIM;
  }
  const float* __restrict__ brow = Wv + (size_t)n * K_VOC;

  v8f acc[4] = {};
  #pragma unroll 2
  for (int k0 = 0; k0 < H_DIM; k0 += 32) {          // A from h_new
    v16bf b;
    cvt8_nt(b, 0, brow + k0 + koff);
    cvt8_nt(b, 8, brow + k0 + 16 + koff);
    #pragma unroll
    for (int mt = 0; mt < 4; ++mt) {
      v16bf a;
      cvt8(a, 0, ah[mt] + k0 + koff);
      cvt8(a, 8, ah[mt] + k0 + 16 + koff);
      acc[mt] = wmma_bf16(a, b, acc[mt]);
    }
  }
  #pragma unroll 2
  for (int k0 = H_DIM; k0 < K_VOC; k0 += 32) {      // A from ctx
    v16bf b;
    cvt8_nt(b, 0, brow + k0 + koff);
    cvt8_nt(b, 8, brow + k0 + 16 + koff);
    #pragma unroll
    for (int mt = 0; mt < 4; ++mt) {
      v16bf a;
      cvt8(a, 0, ac[mt] + k0 + koff);
      cvt8(a, 8, ac[mt] + k0 + 16 + koff);
      acc[mt] = wmma_bf16(a, b, acc[mt]);
    }
  }

  const float bias = bv[n];
  #pragma unroll
  for (int mt = 0; mt < 4; ++mt) {
    #pragma unroll
    for (int r = 0; r < 8; ++r) {
      const int m = mt * 16 + r + hi8;
      out[O_FINAL + (size_t)m * EXT_V + n] = acc[mt][r] + bias;
    }
  }
}

// =====================================================================
// Kernel 8: p_gen = sigmoid([ctx|h|emb] . W_pgen + b)
// =====================================================================
__global__ __launch_bounds__(256)
void k_pgen(const float* __restrict__ ctx, const float* __restrict__ hnew,
            const int* __restrict__ tok, const float* __restrict__ emb,
            const float* __restrict__ Wp, const float* __restrict__ bp,
            float* __restrict__ out, float* __restrict__ wspg) {
  __shared__ float red[256];
  const int b = blockIdx.x, tid = threadIdx.x;
  const float* erow = emb + (size_t)tok[b] * E_DIM;
  float s = 0.f;
  for (int k = tid; k < H2 + H_DIM + E_DIM; k += 256) {
    float x;
    if (k < H2)               x = ctx[b * H2 + k];
    else if (k < H2 + H_DIM)  x = hnew[b * H_DIM + (k - H2)];
    else                      x = erow[k - H2 - H_DIM];
    s += Wp[k] * x;
  }
  red[tid] = s; __syncthreads();
  for (int st = 128; st > 0; st >>= 1) {
    if (tid < st) red[tid] += red[tid + st];
    __syncthreads();
  }
  if (tid == 0) {
    const float p = 1.f / (1.f + expf(-(red[0] + bp[0])));
    out[O_PGEN + b] = p;
    wspg[b] = p;
  }
}

// =====================================================================
// Kernel 9: per-row max & sum(exp) over V (logits live in final region)
// =====================================================================
__global__ __launch_bounds__(256)
void k_rowstat(const float* __restrict__ logits, float* __restrict__ wmax,
               float* __restrict__ wsum) {
  __shared__ float red[256];
  const int b = blockIdx.x, tid = threadIdx.x;
  const float* r = logits + (size_t)b * EXT_V;
  float m = -INFINITY;
  for (int j = tid; j < V_DIM; j += 256) m = fmaxf(m, r[j]);
  red[tid] = m; __syncthreads();
  for (int s = 128; s > 0; s >>= 1) {
    if (tid < s) red[tid] = fmaxf(red[tid], red[tid + s]);
    __syncthreads();
  }
  m = red[0]; __syncthreads();
  float sum = 0.f;
  for (int j = tid; j < V_DIM; j += 256) sum += expf(r[j] - m);
  red[tid] = sum; __syncthreads();
  for (int s = 128; s > 0; s >>= 1) {
    if (tid < s) red[tid] += red[tid + s];
    __syncthreads();
  }
  if (tid == 0) { wmax[b] = m; wsum[b] = red[0]; }
}

// =====================================================================
// Kernel 10: final[:, :V] = p_gen * softmax(logits); final[:, V:] = 0
// =====================================================================
__global__ __launch_bounds__(256)
void k_final(float* __restrict__ out, const float* __restrict__ pg,
             const float* __restrict__ wmax, const float* __restrict__ wsum) {
  const long idx = (long)blockIdx.x * 256 + threadIdx.x;
  if (idx >= (long)B_N * EXT_V) return;
  const int b = (int)(idx / EXT_V);
  const int j = (int)(idx % EXT_V);
  float* p = out + O_FINAL + idx;
  if (j < V_DIM) *p = pg[b] * expf(*p - wmax[b]) / wsum[b];
  else           *p = 0.f;
}

// =====================================================================
// Kernel 11: scatter-add (1-p_gen)*attn at extended indices
// =====================================================================
__global__ __launch_bounds__(256)
void k_scatter(float* __restrict__ out, const float* __restrict__ pg,
               const int* __restrict__ ext) {
  const int idx = blockIdx.x * 256 + threadIdx.x;   // 0..65535
  const int b = idx >> 10;
  const int col = ext[idx];
  const float add = (1.f - pg[b]) * out[O_ATTN + idx];
  atomicAdd(out + O_FINAL + (size_t)b * EXT_V + col, add);
}

// =====================================================================
extern "C" void kernel_launch(void* const* d_in, const int* in_sizes, int n_in,
                              void* d_out, int out_size, void* d_ws, size_t ws_size,
                              hipStream_t stream) {
  const int*   tok   = (const int*)  d_in[0];
  const float* h0    = (const float*)d_in[1];
  const float* c0    = (const float*)d_in[2];
  const float* enc   = (const float*)d_in[3];
  const float* mask  = (const float*)d_in[4];
  const int*   ext   = (const int*)  d_in[5];
  const float* ctxv  = (const float*)d_in[6];
  const float* cov   = (const float*)d_in[7];
  const float* emb   = (const float*)d_in[8];
  const float* w_ih  = (const float*)d_in[9];
  const float* w_hh  = (const float*)d_in[10];
  const float* b_ih  = (const float*)d_in[11];
  const float* b_hh  = (const float*)d_in[12];
  const float* Wh    = (const float*)d_in[13];
  const float* Ws    = (const float*)d_in[14];
  const float* battn = (const float*)d_in[15];
  const float* wc    = (const float*)d_in[16];
  const float* vv    = (const float*)d_in[17];
  const float* Wv    = (const float*)d_in[18];
  const float* bv    = (const float*)d_in[19];
  const float* Wp    = (const float*)d_in[20];
  const float* bp    = (const float*)d_in[21];

  float* out = (float*)d_out;
  float* ws  = (float*)d_ws;
  float* gates = ws + W_GATES;
  float* decf  = ws + W_DECF;
  float* ebuf  = ws + W_E;
  float* pgbuf = ws + W_PG;
  float* mbuf  = ws + W_MAX;
  float* sbuf  = ws + W_SUM;

  k_gates<<<256, 64, 0, stream>>>(tok, emb, ctxv, h0, w_ih, w_hh, b_ih, b_hh, gates);
  k_lstm_point<<<128, 256, 0, stream>>>(gates, c0, out);
  k_decfeat<<<256, 256, 0, stream>>>(out + O_H, Ws, decf);
  k_attn_score<<<4096, 256, 0, stream>>>(enc, Wh, decf, cov, wc, battn, vv, mask, ebuf);
  k_softmax_attn<<<64, 256, 0, stream>>>(ebuf, cov, out);
  k_context<<<256, 256, 0, stream>>>(out + O_ATTN, enc, out);
  k_vocab<<<1563, 64, 0, stream>>>(out + O_H, out + O_CTX, Wv, bv, out);
  k_pgen<<<64, 256, 0, stream>>>(out + O_CTX, out + O_H, tok, emb, Wp, bp, out, pgbuf);
  k_rowstat<<<64, 256, 0, stream>>>(out + O_FINAL, mbuf, sbuf);
  k_final<<<(B_N * EXT_V + 255) / 256, 256, 0, stream>>>(out, pgbuf, mbuf, sbuf);
  k_scatter<<<256, 256, 0, stream>>>(out, pgbuf, ext);
}
